// EGNNLayer_44521630991103
// MI455X (gfx1250) — matrix operand order repaired and checked by reading, not compiled, not run on hardware
//
#include <hip/hip_runtime.h>
#include <hip/hip_bf16.h>

#define HID 128

typedef __attribute__((ext_vector_type(16))) __bf16 v16bf;
typedef __attribute__((ext_vector_type(8)))  float  v8f;
typedef __attribute__((ext_vector_type(4)))  unsigned u32x4;
typedef __attribute__((ext_vector_type(8)))  int      i32x8;
typedef __attribute__((ext_vector_type(4)))  int      i32x4;

union ABu { unsigned u[8]; v16bf v; };

__device__ __forceinline__ float silu_f(float x) {
  return x / (1.0f + __expf(-x));
}

// A fragment (16x32 bf16) from row-major LDS tile, per CDNA5 ISA 16-bit A layout.
__device__ __forceinline__ v16bf load_a_lds(const __bf16* base, int stride, int kt, int lane) {
  const int m  = lane & 15;
  const int kh = lane >> 4;
  const __bf16* r = base + m * stride + kt * 32 + kh * 8;
  ABu ab;
#pragma unroll
  for (int j = 0; j < 4; ++j) ab.u[j]     = *(const unsigned*)(r + 2 * j);
#pragma unroll
  for (int j = 0; j < 4; ++j) ab.u[4 + j] = *(const unsigned*)(r + 16 + 2 * j);
  return ab.v;
}

// B fragment (32x16 bf16) from pre-swizzled pack staged in LDS: 32B contiguous per lane.
__device__ __forceinline__ v16bf load_b_lds(const __bf16* w, int tileIdx, int lane) {
  return *(const v16bf*)(w + (((long)tileIdx * 32 + lane) << 4));
}

// ---------------------------------------------------------------------------
// TDM: stage a contiguous byte range from global into LDS (weights live at
// dynamic-LDS offset 0; kernels use only dynamic LDS).
// D# per CDNA5 ISA ch.8: 1 x (nbytes/8) tile of 8-byte elements.
// ---------------------------------------------------------------------------
__device__ __forceinline__ void tdm_stage_weights(const void* gsrc, unsigned lds_off, unsigned nbytes) {
#if __has_builtin(__builtin_amdgcn_tensor_load_to_lds)
  const unsigned long long ga = (unsigned long long)gsrc;
  const unsigned n8 = nbytes >> 3;                 // 8-byte elements, < 65536
  u32x4 g0;
  g0[0] = 1u;                                      // count=1 (user descriptor)
  g0[1] = lds_off;                                 // lds_addr
  g0[2] = (unsigned)(ga & 0xFFFFFFFFu);            // global_addr[31:0]
  g0[3] = (unsigned)((ga >> 32) & 0x01FFFFFFu) | (2u << 30);  // addr[56:32], type=2
  i32x8 g1;
  g1[0] = (int)(3u << 16);                         // data_size = 8B; no multicast
  g1[1] = (int)((n8 & 0xFFFFu) << 16);             // tensor_dim0[15:0]
  g1[2] = (int)(((n8 >> 16) & 0xFFFFu) | (1u << 16)); // tensor_dim0[31:16], tensor_dim1=1
  g1[3] = (int)((n8 & 0xFFFFu) << 16);             // tile_dim0 = n8
  g1[4] = 1;                                       // tile_dim1 = 1
  g1[5] = (int)n8;                                 // tensor_dim0_stride[31:0]
  g1[6] = (int)((n8 >> 16) & 0xFFFFu);             // tensor_dim0_stride[47:32]
  g1[7] = 0;
  i32x4 gz = {0, 0, 0, 0};
#if defined(__clang_major__) && __clang_major__ >= 23
  i32x8 gz8 = {0, 0, 0, 0, 0, 0, 0, 0};
  __builtin_amdgcn_tensor_load_to_lds(g0, g1, gz, gz, gz8, 0);
#else
  __builtin_amdgcn_tensor_load_to_lds(g0, g1, gz, gz, 0);
#endif
  __builtin_amdgcn_s_wait_tensorcnt(0);
#else
  (void)gsrc; (void)lds_off; (void)nbytes;
#endif
}

// ---------------------------------------------------------------------------
// Weight pre-swizzle: W [Ksrc x 128] f32 row-major -> bf16 B-fragment tiles.
// ---------------------------------------------------------------------------
__global__ void pack_w_kernel(const float* __restrict__ W, unsigned short* __restrict__ dst,
                              int Ksrc, int total) {
  int idx = blockIdx.x * blockDim.x + threadIdx.x;
  if (idx >= total) return;
  const int j    = idx & 15;
  const int lane = (idx >> 4) & 31;
  const int t    = idx >> 9;
  const int nt   = t & 7;
  const int kt   = t >> 3;
  const int K = kt * 32 + ((lane >> 4) << 3) + (j & 7) + ((j >> 3) << 4);
  const int n = nt * 16 + (lane & 15);
  const float v = (K < Ksrc) ? W[K * HID + n] : 0.0f;
  reinterpret_cast<__bf16*>(dst)[idx] = (__bf16)v;
}

__global__ void zero_f32_kernel(float* __restrict__ p, long long n) {
  long long i = (long long)blockIdx.x * blockDim.x + threadIdx.x;
  if (i < n) p[i] = 0.0f;
}

__global__ void copy_f32_kernel(const float* __restrict__ s, float* __restrict__ d, long long n) {
  long long i = (long long)blockIdx.x * blockDim.x + threadIdx.x;
  if (i < n) d[i] = s[i];
}

// ---------------------------------------------------------------------------
// Edge kernel: 8 waves/block, one 16-edge tile per wave. All-dynamic LDS:
//   [0, 139264)                weights: eW1p(73728) | eW2p(32768) | cW1p(32768)
//   [139264 + w*22400, ...)    per-wave tiles
// Total 318,464 B < 320 KB.
// ---------------------------------------------------------------------------
static constexpr int WAVES = 8;
static constexpr int FSTR  = 292;  // feat stride (bf16), 288 K + pad, even
static constexpr int MSTR  = 132;  // 128 + 4, even

static constexpr unsigned EDGE_WREG = 139264;
static constexpr unsigned EW2_OFF   = 73728;
static constexpr unsigned CW1_OFF   = 106496;
static constexpr unsigned PW_FEAT = 0;        // 16*292*2 = 9344
static constexpr unsigned PW_M1   = 9344;     // 16*132*2 = 4224
static constexpr unsigned PW_M2   = 13568;    // 4224
static constexpr unsigned PW_T    = 17792;    // 4224 (bf16)
static constexpr unsigned PW_RAD  = 22016;    // 16*4*4 = 256
static constexpr unsigned PW_ROW  = 22272;    // 16*4   = 64
static constexpr unsigned PW_SZ   = 22400;

__global__ __launch_bounds__(32 * WAVES) void egnn_edge_kernel(
    const float* __restrict__ h, const float* __restrict__ pos,
    const float* __restrict__ eattr, const long long* __restrict__ ei,
    const unsigned short* __restrict__ wpack,  // eW1p|eW2p|cW1p contiguous
    const float* __restrict__ eb1, const float* __restrict__ eb2,
    const float* __restrict__ cb1,
    const float* __restrict__ cW2, const float* __restrict__ cb2,
    float* __restrict__ agg, float* __restrict__ pos_out, int Ne)
{
  extern __shared__ char smem[];

  const int wave = threadIdx.x >> 5;
  const int lane = threadIdx.x & 31;

  // ---- stage all edge-phase weights into LDS (TDM, async DMA) ----
#if __has_builtin(__builtin_amdgcn_tensor_load_to_lds)
  if (wave == 0) tdm_stage_weights(wpack, 0u, EDGE_WREG);
#else
  {
    const uint4* s = (const uint4*)wpack;
    uint4* d = (uint4*)smem;
    for (unsigned i = threadIdx.x; i < EDGE_WREG / 16; i += blockDim.x) d[i] = s[i];
  }
#endif

  const __bf16* wE1 = (const __bf16*)(smem);
  const __bf16* wE2 = (const __bf16*)(smem + EW2_OFF);
  const __bf16* wC1 = (const __bf16*)(smem + CW1_OFF);

  char* pw = smem + EDGE_WREG + (unsigned)wave * PW_SZ;
  __bf16* feat = (__bf16*)(pw + PW_FEAT);
  __bf16* m1   = (__bf16*)(pw + PW_M1);
  __bf16* m2   = (__bf16*)(pw + PW_M2);
  __bf16* tt   = (__bf16*)(pw + PW_T);
  float*  rad  = (float*)(pw + PW_RAD);
  int*    rowp = (int*)(pw + PW_ROW);

  const long long tile   = (long long)blockIdx.x * WAVES + wave;
  const long long ntiles = ((long long)Ne + 15) >> 4;
  const bool active = (tile < ntiles);
  const long long ebase = tile << 4;

  // ---- stage 0: gather edge features (bf16) into LDS ----
  if (active) {
    const int e    = lane & 15;
    const int half = lane >> 4;
    const long long eg = ebase + e;
    __bf16* frow = feat + e * FSTR + half * HID;
    if (eg < Ne) {
      const long long node = ei[(long long)half * Ne + eg];
      const float4* hp = (const float4*)(h + node * HID);
#pragma unroll
      for (int i = 0; i < 32; ++i) {
        float4 v = hp[i];
        frow[4 * i + 0] = (__bf16)v.x;
        frow[4 * i + 1] = (__bf16)v.y;
        frow[4 * i + 2] = (__bf16)v.z;
        frow[4 * i + 3] = (__bf16)v.w;
      }
    } else {
#pragma unroll
      for (int i = 0; i < HID; ++i) frow[i] = (__bf16)0.0f;
    }
    if (lane < 16) {
      __bf16* fr = feat + e * FSTR;
      if (eg < Ne) {
        const long long rn_ = ei[eg];
        const long long cn_ = ei[Ne + eg];
        rowp[e] = (int)rn_;
        const float dx = pos[rn_ * 3 + 0] - pos[cn_ * 3 + 0];
        const float dy = pos[rn_ * 3 + 1] - pos[cn_ * 3 + 1];
        const float dz = pos[rn_ * 3 + 2] - pos[cn_ * 3 + 2];
        float rnorm = sqrtf(dx * dx + dy * dy + dz * dz);
        rnorm = fmaxf(rnorm, 1e-8f);
        rad[e * 4 + 0] = dx;
        rad[e * 4 + 1] = dy;
        rad[e * 4 + 2] = dz;
        rad[e * 4 + 3] = rnorm;
        fr[256] = (__bf16)rnorm;
        fr[257] = (__bf16)eattr[eg];
      } else {
        rowp[e] = 0;
        rad[e * 4 + 0] = 0.0f; rad[e * 4 + 1] = 0.0f;
        rad[e * 4 + 2] = 0.0f; rad[e * 4 + 3] = 1.0f;
        fr[256] = (__bf16)0.0f;
        fr[257] = (__bf16)0.0f;
      }
      for (int k = 258; k < 288; ++k) fr[k] = (__bf16)0.0f;  // K pad
    }
  }
  __syncthreads();   // also publishes TDM-staged weights

  const int nl = lane & 15;
  const int mh = (lane >> 4) << 3;

  // ---- GEMM1: m1 = silu(feat @ eW1 + eb1)   (K = 9 tiles of 32) ----
  // kt-outer / nt-inner: load each A fragment once, 8 independent acc chains.
  if (active) {
    v8f acc[8];
#pragma unroll
    for (int nt = 0; nt < 8; ++nt) acc[nt] = (v8f){};
#pragma unroll
    for (int kt = 0; kt < 9; ++kt) {
      v16bf a = load_a_lds(feat, FSTR, kt, lane);
#pragma unroll
      for (int nt = 0; nt < 8; ++nt) {
        v16bf b = load_b_lds(wE1, kt * 8 + nt, lane);
        acc[nt] = __builtin_amdgcn_wmma_f32_16x16x32_bf16(false, a, false, b, (short)0, acc[nt], false, false);
      }
    }
#pragma unroll
    for (int nt = 0; nt < 8; ++nt) {
      const int col = nt * 16 + nl;
      const float bias = eb1[col];
#pragma unroll
      for (int r = 0; r < 8; ++r)
        m1[(mh + r) * MSTR + col] = (__bf16)silu_f(acc[nt][r] + bias);
    }
  }
  __syncthreads();

  // ---- GEMM2: m2 = silu(m1 @ eW2 + eb2); scatter-add into agg ----
  if (active) {
    v8f acc[8];
#pragma unroll
    for (int nt = 0; nt < 8; ++nt) acc[nt] = (v8f){};
#pragma unroll
    for (int kt = 0; kt < 4; ++kt) {
      v16bf a = load_a_lds(m1, MSTR, kt, lane);
#pragma unroll
      for (int nt = 0; nt < 8; ++nt) {
        v16bf b = load_b_lds(wE2, kt * 8 + nt, lane);
        acc[nt] = __builtin_amdgcn_wmma_f32_16x16x32_bf16(false, a, false, b, (short)0, acc[nt], false, false);
      }
    }
#pragma unroll
    for (int nt = 0; nt < 8; ++nt) {
      const int col = nt * 16 + nl;
      const float bias = eb2[col];
#pragma unroll
      for (int r = 0; r < 8; ++r) {
        const int M = mh + r;
        float x = silu_f(acc[nt][r] + bias);
        m2[M * MSTR + col] = (__bf16)x;
        if (ebase + M < Ne)
          atomicAdd(&agg[(long long)rowp[M] * HID + col], x);
      }
    }
  }
  __syncthreads();

  // ---- GEMM3: t = silu(m2 @ cW1 + cb1) ----
  if (active) {
    v8f acc[8];
#pragma unroll
    for (int nt = 0; nt < 8; ++nt) acc[nt] = (v8f){};
#pragma unroll
    for (int kt = 0; kt < 4; ++kt) {
      v16bf a = load_a_lds(m2, MSTR, kt, lane);
#pragma unroll
      for (int nt = 0; nt < 8; ++nt) {
        v16bf b = load_b_lds(wC1, kt * 8 + nt, lane);
        acc[nt] = __builtin_amdgcn_wmma_f32_16x16x32_bf16(false, a, false, b, (short)0, acc[nt], false, false);
      }
    }
#pragma unroll
    for (int nt = 0; nt < 8; ++nt) {
      const int col = nt * 16 + nl;
      const float bias = cb1[col];
#pragma unroll
      for (int r = 0; r < 8; ++r)
        tt[(mh + r) * MSTR + col] = (__bf16)silu_f(acc[nt][r] + bias);
    }
  }
  __syncthreads();

  // ---- coord_diff = t @ cW2 + cb2 ; scatter coord update ----
  if (active && lane < 16) {
    const long long eg = ebase + lane;
    if (eg < Ne) {
      float dot = cb2[0];
#pragma unroll 8
      for (int k = 0; k < HID; ++k) dot += (float)tt[lane * MSTR + k] * cW2[k];
      const float inv = 1.0f / rad[lane * 4 + 3];
      const long long rn_ = rowp[lane];
#pragma unroll
      for (int c = 0; c < 3; ++c)
        atomicAdd(&pos_out[rn_ * 3 + c], dot * rad[lane * 4 + c] * inv);
    }
  }
}

// ---------------------------------------------------------------------------
// Node kernel: h_new = h + silu([h | agg] @ nW1 + nb1) @ nW2 + nb2
// LDS: [0, 98304) weights nW1p(65536)|nW2p(32768); per-wave 12800 B. Tot 200,704.
// ---------------------------------------------------------------------------
static constexpr int NSTR = 260;  // 256 + 4, even
static constexpr unsigned NODE_WREG = 98304;
static constexpr unsigned NW2_OFF   = 65536;
static constexpr unsigned NPW_FEAT  = 0;      // 16*260*2 = 8320
static constexpr unsigned NPW_M1    = 8320;   // 16*132*2 = 4224
static constexpr unsigned NPW_SZ    = 12800;

__global__ __launch_bounds__(32 * WAVES) void egnn_node_kernel(
    const float* __restrict__ h, const float* __restrict__ agg,
    const unsigned short* __restrict__ wpack,  // nW1p|nW2p contiguous
    const float* __restrict__ nb1, const float* __restrict__ nb2,
    float* __restrict__ out_h, int Nn)
{
  extern __shared__ char smem[];

  const int wave = threadIdx.x >> 5;
  const int lane = threadIdx.x & 31;

#if __has_builtin(__builtin_amdgcn_tensor_load_to_lds)
  if (wave == 0) tdm_stage_weights(wpack, 0u, NODE_WREG);
#else
  {
    const uint4* s = (const uint4*)wpack;
    uint4* d = (uint4*)smem;
    for (unsigned i = threadIdx.x; i < NODE_WREG / 16; i += blockDim.x) d[i] = s[i];
  }
#endif

  const __bf16* wN1 = (const __bf16*)(smem);
  const __bf16* wN2 = (const __bf16*)(smem + NW2_OFF);

  char* pw = smem + NODE_WREG + (unsigned)wave * NPW_SZ;
  __bf16* feat = (__bf16*)(pw + NPW_FEAT);
  __bf16* m1   = (__bf16*)(pw + NPW_M1);

  const long long tile   = (long long)blockIdx.x * WAVES + wave;
  const long long ntiles = ((long long)Nn + 15) >> 4;
  const bool active = (tile < ntiles);
  const long long nbase = tile << 4;

  // gather [h | agg]
  if (active) {
    const int e    = lane & 15;
    const int half = lane >> 4;
    const long long node = nbase + e;
    __bf16* frow = feat + e * NSTR + half * HID;
    if (node < Nn) {
      const float4* sp = (const float4*)((half ? agg : h) + node * HID);
#pragma unroll
      for (int i = 0; i < 32; ++i) {
        float4 v = sp[i];
        frow[4 * i + 0] = (__bf16)v.x;
        frow[4 * i + 1] = (__bf16)v.y;
        frow[4 * i + 2] = (__bf16)v.z;
        frow[4 * i + 3] = (__bf16)v.w;
      }
    } else {
#pragma unroll
      for (int i = 0; i < HID; ++i) frow[i] = (__bf16)0.0f;
    }
  }
  __syncthreads();   // also publishes TDM-staged weights

  const int nl = lane & 15;
  const int mh = (lane >> 4) << 3;

  // GEMM1: silu(feat @ nW1 + nb1)  (K = 8 tiles, exact)
  if (active) {
    v8f acc[8];
#pragma unroll
    for (int nt = 0; nt < 8; ++nt) acc[nt] = (v8f){};
#pragma unroll
    for (int kt = 0; kt < 8; ++kt) {
      v16bf a = load_a_lds(feat, NSTR, kt, lane);
#pragma unroll
      for (int nt = 0; nt < 8; ++nt) {
        v16bf b = load_b_lds(wN1, kt * 8 + nt, lane);
        acc[nt] = __builtin_amdgcn_wmma_f32_16x16x32_bf16(false, a, false, b, (short)0, acc[nt], false, false);
      }
    }
#pragma unroll
    for (int nt = 0; nt < 8; ++nt) {
      const int col = nt * 16 + nl;
      const float bias = nb1[col];
#pragma unroll
      for (int r = 0; r < 8; ++r)
        m1[(mh + r) * MSTR + col] = (__bf16)silu_f(acc[nt][r] + bias);
    }
  }
  __syncthreads();

  // GEMM2 + residual (no activation on output layer)
  if (active) {
    v8f acc[8];
#pragma unroll
    for (int nt = 0; nt < 8; ++nt) acc[nt] = (v8f){};
#pragma unroll
    for (int kt = 0; kt < 4; ++kt) {
      v16bf a = load_a_lds(m1, MSTR, kt, lane);
#pragma unroll
      for (int nt = 0; nt < 8; ++nt) {
        v16bf b = load_b_lds(wN2, kt * 8 + nt, lane);
        acc[nt] = __builtin_amdgcn_wmma_f32_16x16x32_bf16(false, a, false, b, (short)0, acc[nt], false, false);
      }
    }
#pragma unroll
    for (int nt = 0; nt < 8; ++nt) {
      const int col = nt * 16 + nl;
      const float bias = nb2[col];
#pragma unroll
      for (int r = 0; r < 8; ++r) {
        const long long node = nbase + mh + r;
        if (node < Nn)
          out_h[node * HID + col] = h[node * HID + col] + acc[nt][r] + bias;
      }
    }
  }
}

// ---------------------------------------------------------------------------
extern "C" void kernel_launch(void* const* d_in, const int* in_sizes, int n_in,
                              void* d_out, int out_size, void* d_ws, size_t ws_size,
                              hipStream_t stream) {
  const float* h     = (const float*)d_in[0];
  const float* pos   = (const float*)d_in[1];
  const float* eattr = (const float*)d_in[2];
  const long long* ei = (const long long*)d_in[3];
  const float* eW1 = (const float*)d_in[4];
  const float* eb1 = (const float*)d_in[5];
  const float* eW2 = (const float*)d_in[6];
  const float* eb2 = (const float*)d_in[7];
  const float* nW1 = (const float*)d_in[8];
  const float* nb1 = (const float*)d_in[9];
  const float* nW2 = (const float*)d_in[10];
  const float* nb2 = (const float*)d_in[11];
  const float* cW1 = (const float*)d_in[12];
  const float* cb1 = (const float*)d_in[13];
  const float* cW2 = (const float*)d_in[14];
  const float* cb2 = (const float*)d_in[15];

  const int Nn = in_sizes[0] / HID;
  const int Ne = in_sizes[3] / 2;

  float* out_h   = (float*)d_out;
  float* out_pos = out_h + (size_t)Nn * HID;

  // workspace: agg | eW1p | eW2p | cW1p | nW1p | nW2p  (packs contiguous)
  char* ws = (char*)d_ws;
  float* agg = (float*)ws;
  size_t off = (size_t)Nn * HID * sizeof(float);
  unsigned short* eW1p = (unsigned short*)(ws + off); off += (size_t)9 * 8 * 512 * 2;  // 73728
  unsigned short* eW2p = (unsigned short*)(ws + off); off += (size_t)4 * 8 * 512 * 2;  // 32768
  unsigned short* cW1p = (unsigned short*)(ws + off); off += (size_t)4 * 8 * 512 * 2;  // 32768
  unsigned short* nW1p = (unsigned short*)(ws + off); off += (size_t)8 * 8 * 512 * 2;  // 65536
  unsigned short* nW2p = (unsigned short*)(ws + off); off += (size_t)4 * 8 * 512 * 2;  // 32768

  // weight pre-swizzle to WMMA B-fragment layout (bf16)
  pack_w_kernel<<<(9 * 8 * 512 + 255) / 256, 256, 0, stream>>>(eW1, eW1p, 258, 9 * 8 * 512);
  pack_w_kernel<<<(4 * 8 * 512 + 255) / 256, 256, 0, stream>>>(eW2, eW2p, 128, 4 * 8 * 512);
  pack_w_kernel<<<(4 * 8 * 512 + 255) / 256, 256, 0, stream>>>(cW1, cW1p, 128, 4 * 8 * 512);
  pack_w_kernel<<<(8 * 8 * 512 + 255) / 256, 256, 0, stream>>>(nW1, nW1p, 256, 8 * 8 * 512);
  pack_w_kernel<<<(4 * 8 * 512 + 255) / 256, 256, 0, stream>>>(nW2, nW2p, 128, 4 * 8 * 512);

  const long long aggN = (long long)Nn * HID;
  zero_f32_kernel<<<(unsigned)((aggN + 255) / 256), 256, 0, stream>>>(agg, aggN);
  copy_f32_kernel<<<(Nn * 3 + 255) / 256, 256, 0, stream>>>(pos, out_pos, (long long)Nn * 3);

  const int etiles = (Ne + 15) / 16;
  const int eblocks = (etiles + WAVES - 1) / WAVES;
  const unsigned edge_lds = EDGE_WREG + WAVES * PW_SZ;  // 318,464 B
  egnn_edge_kernel<<<eblocks, 32 * WAVES, edge_lds, stream>>>(
      h, pos, eattr, ei, eW1p, eb1, eb2, cb1, cW2, cb2, agg, out_pos, Ne);

  const int ntile = (Nn + 15) / 16;
  const int nblocks = (ntile + WAVES - 1) / WAVES;
  const unsigned node_lds = NODE_WREG + WAVES * NPW_SZ;  // 200,704 B
  egnn_node_kernel<<<nblocks, 32 * WAVES, node_lds, stream>>>(
      h, agg, nW1p, nb1, nb2, out_h, Nn);
}
